// GCN_24988119728417
// MI455X (gfx1250) — matrix operand order repaired and checked by reading, not compiled, not run on hardware
//
#include <hip/hip_runtime.h>
#include <cstdint>
#include <cstddef>

// ---------------------------------------------------------------------------
// MI455X (gfx1250) GNN forward.  All GEMMs on v_wmma_f32_16x16x32_f16.
// wave32; 256 threads = 8 waves per block; each block owns 64 rows.
// Uniform weight/tile slabs are fetched with the Tensor Data Mover
// (tensor_load_to_lds, TENSORcnt) and double-buffered in LDS.
// ---------------------------------------------------------------------------

typedef _Float16 v16h  __attribute__((ext_vector_type(16)));
typedef _Float16 half8 __attribute__((ext_vector_type(8)));
typedef float    v8f   __attribute__((ext_vector_type(8)));

#define DEV static __device__ __forceinline__

#if __has_builtin(__builtin_amdgcn_tensor_load_to_lds)
#define USE_TDM 1
#else
#define USE_TDM 0
#endif

#if USE_TDM
typedef unsigned int tdm_u32x4 __attribute__((ext_vector_type(4)));
typedef int          tdm_i32x8 __attribute__((ext_vector_type(8)));
typedef int          tdm_i32x4 __attribute__((ext_vector_type(4)));

// TDM load of a 2D f16 tile: nrows rows x 32 elements, row stride
// row_stride_elems, into LDS at byte offset lds_off.  D# per ISA 8.3/8.4:
//  g0: count=1 | lds_addr | global_addr[56:0] | type=2
//  g1: data_size=2B, tensor_dim0=32, tensor_dim1=nrows, tile_dim0=32,
//      tile_dim1=nrows, tensor_dim0_stride=row_stride_elems
DEV void tdm_load_2d_f16(const _Float16* gsrc, unsigned lds_off,
                         int nrows, int row_stride_elems)
{
  unsigned long long ga = (unsigned long long)(uintptr_t)gsrc;
  tdm_u32x4 g0 = { 1u,                                   // count = 1 (valid)
                   lds_off,                              // lds_addr
                   (unsigned)ga,                         // global_addr lo
                   (unsigned)((ga >> 32) & 0x01FFFFFFu)  // global_addr hi
                   | (2u << 30) };                       // type = 2 (image)
  tdm_i32x8 g1 = { (int)0x00010000u,                     // data_size = 2B
                   (int)(32u << 16),                     // tensor_dim0 lo16
                   (int)((unsigned)nrows << 16),         // dim0 hi | tensor_dim1 lo
                   (int)(32u << 16),                     // dim1 hi | tile_dim0 = 32
                   (int)(unsigned)nrows,                 // tile_dim1 | tile_dim2 = 0
                   (int)(unsigned)row_stride_elems,      // tensor_dim0_stride lo32
                   0, 0 };
  tdm_i32x4 gz = { 0, 0, 0, 0 };
#if defined(__clang_major__) && __clang_major__ >= 23
  tdm_i32x8 gz8 = { 0, 0, 0, 0, 0, 0, 0, 0 };
  __builtin_amdgcn_tensor_load_to_lds(g0, g1, gz, gz, gz8, 0);
#else
  __builtin_amdgcn_tensor_load_to_lds(g0, g1, gz, gz, 0);
#endif
}

DEV unsigned lds_off_of(const void* p) {   // generic LDS addr: low 32 bits = offset
  return (unsigned)(uintptr_t)p;
}
#endif  // USE_TDM

// A operand (16x32 f16, MxK).  ISA 7.12.2: lanes 0-15 hold M=lane with
// K=[8h..8h+7] in v0-3 and K=[16+8h..16+8h+7] in v4-7 (h = lane>>4).
DEV v16h load_a16(const _Float16* rowptr, int h) {
  half8 lo = *(const half8*)(rowptr + 8 * h);
  half8 hi = *(const half8*)(rowptr + 16 + 8 * h);
  v16h a;
#pragma unroll
  for (int i = 0; i < 8; ++i) { a[i] = lo[i]; a[8 + i] = hi[i]; }
  return a;
}

// B operand (32x16 f16, KxN) from K-transposed storage Wt[n][k0..k0+31]:
// lane holds N=lane&15, K=[8h..8h+7] then K=[16+8h..16+8h+7] (same as A).
DEV v16h load_b16(const _Float16* colptr, int h) {
  half8 lo = *(const half8*)(colptr + 8 * h);
  half8 hi = *(const half8*)(colptr + 16 + 8 * h);
  v16h b;
#pragma unroll
  for (int i = 0; i < 8; ++i) { b[i] = lo[i]; b[8 + i] = hi[i]; }
  return b;
}

enum { MODE_NODE_ENC = 0, MODE_EDGE_ENC = 1, MODE_MSG = 2, MODE_NODE_UPD = 3, MODE_OUT = 4 };

// Fused 2-layer MLP:  O = relu(A @ W1 + b1) @ W2 + b2, 64 rows per block.
// W1t: [N1][K1P] f16 (K-transposed, zero-padded), W2t: [N2][N1] f16.
template <int MODE>
__global__ __launch_bounds__(256) void mlp2_kernel(
    const float*    __restrict__ Af32,
    const _Float16* __restrict__ Af16,
    const _Float16* __restrict__ gather_h,
    const _Float16* __restrict__ gather_e,
    const int*      __restrict__ src_idx,
    const int*      __restrict__ dst_idx,
    const _Float16* __restrict__ W1t,
    const float*    __restrict__ b1,
    const _Float16* __restrict__ W2t,
    const float*    __restrict__ b2,
    float*          __restrict__ out_f32,
    _Float16*       __restrict__ out_f16,
    float*          __restrict__ agg,
    int M)
{
  constexpr int K1  = (MODE == MODE_NODE_ENC) ? 16 : (MODE == MODE_EDGE_ENC) ? 8
                    : (MODE == MODE_MSG) ? 512 : 256;
  constexpr int K1P = (K1 < 32) ? 32 : K1;
  constexpr int N1  = (MODE == MODE_OUT) ? 128 : 256;
  constexpr int N2  = N1;
  constexpr int NT1 = N1 / 32;   // 16x16 output tiles per wave (2 waves per mtile)
  constexpr int NT2 = N2 / 32;
  constexpr int NK1 = K1P / 32;
  constexpr int NK2 = N1 / 32;

  __shared__ __attribute__((aligned(64))) _Float16 ldsA[64 * 32];
  __shared__ __attribute__((aligned(64))) _Float16 ldsW[2][N1 * 32];
  __shared__ __attribute__((aligned(64))) _Float16 ldsT[64 * N1];

  const int tid   = threadIdx.x;
  const int lane  = tid & 31;
  const int wave  = tid >> 5;
  const int mtile = wave & 3;    // 4 row tiles of 16
  const int pair  = wave >> 2;   // 2 waves per mtile split the N tiles
  const int h16   = lane >> 4;
  const int nlane = lane & 15;
  const int rowbase = blockIdx.x * 64;

  const v8f vzero = {0.f, 0.f, 0.f, 0.f, 0.f, 0.f, 0.f, 0.f};
  v8f acc[NT1];
#pragma unroll
  for (int j = 0; j < NT1; ++j) acc[j] = vzero;

  // ---------------- stage 1: T = relu(A @ W1 + b1) ----------------
#if USE_TDM
  if (wave == 0) tdm_load_2d_f16(W1t, lds_off_of(&ldsW[0][0]), N1, K1P);
#endif
  for (int ks = 0; ks < NK1; ++ks) {
    const int k0 = ks * 32;
    {   // A slab [64][32] -> LDS (one 16B chunk per thread)
      int r  = tid >> 2;
      int kc = (tid & 3) * 8;
      int row = rowbase + r; if (row >= M) row = M - 1;
      int k = k0 + kc;
      half8 v;
      if (MODE == MODE_MSG) {
        if (k < 256) { int s = src_idx[row]; v = *(const half8*)(gather_h + s * 256 + k); }
        else         { v = *(const half8*)(gather_e + row * 256 + (k - 256)); }
      } else if (MODE == MODE_OUT) {
        v = *(const half8*)(Af16 + row * K1 + k);
      } else if (K1P == K1) {      // full-width f32 source (agg)
        const float4 f0 = *(const float4*)(Af32 + row * K1 + k);
        const float4 f1 = *(const float4*)(Af32 + row * K1 + k + 4);
        v[0] = (_Float16)f0.x; v[1] = (_Float16)f0.y; v[2] = (_Float16)f0.z; v[3] = (_Float16)f0.w;
        v[4] = (_Float16)f1.x; v[5] = (_Float16)f1.y; v[6] = (_Float16)f1.z; v[7] = (_Float16)f1.w;
      } else {                     // narrow f32 source, zero-pad K to 32
#pragma unroll
        for (int j = 0; j < 8; ++j) {
          float f = (k + j < K1) ? Af32[row * K1 + k + j] : 0.f;
          v[j] = (_Float16)f;
        }
      }
      *(half8*)(ldsA + r * 32 + kc) = v;
    }
#if USE_TDM
    if (wave == 0) {   // prefetch next W slab with the TDM, wait for current
      if (ks + 1 < NK1) {
        tdm_load_2d_f16(W1t + (ks + 1) * 32, lds_off_of(&ldsW[(ks + 1) & 1][0]), N1, K1P);
        __builtin_amdgcn_s_wait_tensorcnt(1);
      } else {
        __builtin_amdgcn_s_wait_tensorcnt(0);
      }
    }
#else
    for (int n = tid; n < N1; n += 256) {   // W1 slab [N1][32] (64B per row)
      const uint4* s = (const uint4*)(W1t + n * K1P + k0);
      uint4* d = (uint4*)(&ldsW[ks & 1][n * 32]);
      d[0] = s[0]; d[1] = s[1]; d[2] = s[2]; d[3] = s[3];
    }
#endif
    __syncthreads();
    v16h a = load_a16(ldsA + (mtile * 16 + nlane) * 32, h16);
#pragma unroll
    for (int j = 0; j < NT1; ++j) {
      v16h b = load_b16(&ldsW[ks & 1][((pair + 2 * j) * 16 + nlane) * 32], h16);
      acc[j] = __builtin_amdgcn_wmma_f32_16x16x32_f16(false, a, false, b, (short)0,
                                                      acc[j], false, false);
    }
    __syncthreads();
  }

#if USE_TDM
  // prologue for stage 2 (overlaps the ReLU/T-store phase; ldsW[0] is free
  // because every wave passed the trailing barrier of stage 1)
  if (wave == 0) tdm_load_2d_f16(W2t, lds_off_of(&ldsW[0][0]), N2, N1);
#endif

  // bias + ReLU -> f16 intermediate in LDS (C layout: M = i + 8*h16, N = nlane)
#pragma unroll
  for (int j = 0; j < NT1; ++j) {
    int n = (pair + 2 * j) * 16 + nlane;
    float bias = b1[n];
#pragma unroll
    for (int i = 0; i < 8; ++i) {
      float v = acc[j][i] + bias;
      v = v > 0.f ? v : 0.f;
      ldsT[(mtile * 16 + i + 8 * h16) * N1 + n] = (_Float16)v;
    }
  }
  __syncthreads();

  // ---------------- stage 2: O = T @ W2 + b2 ----------------
#pragma unroll
  for (int j = 0; j < NT2; ++j) acc[j] = vzero;
  for (int ks = 0; ks < NK2; ++ks) {
    const int k0 = ks * 32;
#if USE_TDM
    if (wave == 0) {
      if (ks + 1 < NK2) {
        tdm_load_2d_f16(W2t + (ks + 1) * 32, lds_off_of(&ldsW[(ks + 1) & 1][0]), N2, N1);
        __builtin_amdgcn_s_wait_tensorcnt(1);
      } else {
        __builtin_amdgcn_s_wait_tensorcnt(0);
      }
    }
#else
    for (int n = tid; n < N2; n += 256) {
      const uint4* s = (const uint4*)(W2t + n * N1 + k0);
      uint4* d = (uint4*)(&ldsW[ks & 1][n * 32]);
      d[0] = s[0]; d[1] = s[1]; d[2] = s[2]; d[3] = s[3];
    }
#endif
    __syncthreads();
    v16h a = load_a16(ldsT + (mtile * 16 + nlane) * N1 + k0, h16);
#pragma unroll
    for (int j = 0; j < NT2; ++j) {
      v16h b = load_b16(&ldsW[ks & 1][((pair + 2 * j) * 16 + nlane) * 32], h16);
      acc[j] = __builtin_amdgcn_wmma_f32_16x16x32_f16(false, a, false, b, (short)0,
                                                      acc[j], false, false);
    }
    __syncthreads();
  }

  // ---------------- epilogue ----------------
#pragma unroll
  for (int j = 0; j < NT2; ++j) {
    int n = (pair + 2 * j) * 16 + nlane;
    float bias = b2[n];
#pragma unroll
    for (int i = 0; i < 8; ++i) {
      int row = rowbase + mtile * 16 + i + 8 * h16;
      if (row >= M) continue;
      float v = acc[j][i] + bias;
      if (MODE == MODE_MSG) {            // scatter-sum into node accumulator
        int drow = dst_idx[row];
        __hip_atomic_fetch_add(&agg[drow * 256 + n], v,
                               __ATOMIC_RELAXED, __HIP_MEMORY_SCOPE_AGENT);
      } else if (MODE == MODE_NODE_UPD) {  // h = O + h (residual), keep f32+f16
        float nv = v + out_f32[row * N2 + n];
        out_f32[row * N2 + n] = nv;
        out_f16[row * N2 + n] = (_Float16)nv;
      } else if (MODE == MODE_NODE_ENC) {
        out_f32[row * N2 + n] = v;
        out_f16[row * N2 + n] = (_Float16)v;
      } else {                             // EDGE_ENC / OUT: f16 only
        out_f16[row * N2 + n] = (_Float16)v;
      }
    }
  }
}

// Single linear layer (encoder): out = A @ W + b, f16 in/out.
// Both the A tile and the W slab are uniform 2D tiles -> TDM + double buffer.
template <int K, int N>
__global__ __launch_bounds__(256) void linear_kernel(
    const _Float16* __restrict__ A, const _Float16* __restrict__ Wt,
    const float* __restrict__ bias, _Float16* __restrict__ out, int M)
{
  constexpr int NT = N / 32;
  constexpr int NK = K / 32;
  __shared__ __attribute__((aligned(64))) _Float16 ldsA[2][64 * 32];
  __shared__ __attribute__((aligned(64))) _Float16 ldsW[2][N * 32];

  const int tid = threadIdx.x, lane = tid & 31, wave = tid >> 5;
  const int mtile = wave & 3, pair = wave >> 2;
  const int h16 = lane >> 4, nlane = lane & 15;
  const int rowbase = blockIdx.x * 64;

  const v8f vzero = {0.f, 0.f, 0.f, 0.f, 0.f, 0.f, 0.f, 0.f};
  v8f acc[NT];
#pragma unroll
  for (int j = 0; j < NT; ++j) acc[j] = vzero;

#if USE_TDM
  if (wave == 0) {
    tdm_load_2d_f16(A + (size_t)rowbase * K, lds_off_of(&ldsA[0][0]), 64, K);
    tdm_load_2d_f16(Wt, lds_off_of(&ldsW[0][0]), N, K);
  }
#endif
  for (int ks = 0; ks < NK; ++ks) {
    const int k0 = ks * 32;
#if USE_TDM
    if (wave == 0) {
      if (ks + 1 < NK) {
        tdm_load_2d_f16(A + (size_t)rowbase * K + (ks + 1) * 32,
                        lds_off_of(&ldsA[(ks + 1) & 1][0]), 64, K);
        tdm_load_2d_f16(Wt + (ks + 1) * 32, lds_off_of(&ldsW[(ks + 1) & 1][0]), N, K);
        __builtin_amdgcn_s_wait_tensorcnt(2);
      } else {
        __builtin_amdgcn_s_wait_tensorcnt(0);
      }
    }
#else
    {
      int r = tid >> 2, kc = (tid & 3) * 8;
      int row = rowbase + r; if (row >= M) row = M - 1;
      *(half8*)(&ldsA[ks & 1][r * 32 + kc]) = *(const half8*)(A + row * K + k0 + kc);
    }
    for (int n = tid; n < N; n += 256) {
      const uint4* s = (const uint4*)(Wt + n * K + k0);
      uint4* d = (uint4*)(&ldsW[ks & 1][n * 32]);
      d[0] = s[0]; d[1] = s[1]; d[2] = s[2]; d[3] = s[3];
    }
#endif
    __syncthreads();
    v16h a = load_a16(&ldsA[ks & 1][(mtile * 16 + nlane) * 32], h16);
#pragma unroll
    for (int j = 0; j < NT; ++j) {
      v16h b = load_b16(&ldsW[ks & 1][((pair + 2 * j) * 16 + nlane) * 32], h16);
      acc[j] = __builtin_amdgcn_wmma_f32_16x16x32_f16(false, a, false, b, (short)0,
                                                      acc[j], false, false);
    }
    __syncthreads();
  }
#pragma unroll
  for (int j = 0; j < NT; ++j) {
    int n = (pair + 2 * j) * 16 + nlane;
    float bv = bias[n];
#pragma unroll
    for (int i = 0; i < 8; ++i) {
      int row = rowbase + mtile * 16 + i + 8 * h16;
      if (row < M) out[row * N + n] = (_Float16)(acc[j][i] + bv);
    }
  }
}

// out[4096,4096] = henc[4096,128] @ hf[4096,128]^T.  hf rows ARE the
// K-transposed B layout; both slabs are uniform tiles -> TDM + double buffer.
__global__ __launch_bounds__(256) void hht_kernel(
    const _Float16* __restrict__ henc, const _Float16* __restrict__ hf,
    float* __restrict__ out)
{
  __shared__ __attribute__((aligned(64))) _Float16 ldsA[2][64 * 32];
  __shared__ __attribute__((aligned(64))) _Float16 ldsB[2][64 * 32];

  const int tid = threadIdx.x, lane = tid & 31, wave = tid >> 5;
  const int mtile = wave & 3, pair = wave >> 2;
  const int h16 = lane >> 4, nlane = lane & 15;
  const int rowbase = blockIdx.y * 64;
  const int colbase = blockIdx.x * 64;

  const v8f vzero = {0.f, 0.f, 0.f, 0.f, 0.f, 0.f, 0.f, 0.f};
  v8f acc[2] = {vzero, vzero};

#if USE_TDM
  if (wave == 0) {
    tdm_load_2d_f16(henc + (size_t)rowbase * 128, lds_off_of(&ldsA[0][0]), 64, 128);
    tdm_load_2d_f16(hf   + (size_t)colbase * 128, lds_off_of(&ldsB[0][0]), 64, 128);
  }
#endif
  for (int ks = 0; ks < 4; ++ks) {
    const int k0 = ks * 32;
#if USE_TDM
    if (wave == 0) {
      if (ks + 1 < 4) {
        tdm_load_2d_f16(henc + (size_t)rowbase * 128 + (ks + 1) * 32,
                        lds_off_of(&ldsA[(ks + 1) & 1][0]), 64, 128);
        tdm_load_2d_f16(hf + (size_t)colbase * 128 + (ks + 1) * 32,
                        lds_off_of(&ldsB[(ks + 1) & 1][0]), 64, 128);
        __builtin_amdgcn_s_wait_tensorcnt(2);
      } else {
        __builtin_amdgcn_s_wait_tensorcnt(0);
      }
    }
#else
    {
      int r = tid >> 2, kc = (tid & 3) * 8;
      *(half8*)(&ldsA[ks & 1][r * 32 + kc]) = *(const half8*)(henc + (rowbase + r) * 128 + k0 + kc);
      *(half8*)(&ldsB[ks & 1][r * 32 + kc]) = *(const half8*)(hf   + (colbase + r) * 128 + k0 + kc);
    }
#endif
    __syncthreads();
    v16h a = load_a16(&ldsA[ks & 1][(mtile * 16 + nlane) * 32], h16);
#pragma unroll
    for (int j = 0; j < 2; ++j) {
      v16h b = load_b16(&ldsB[ks & 1][((pair + 2 * j) * 16 + nlane) * 32], h16);
      acc[j] = __builtin_amdgcn_wmma_f32_16x16x32_f16(false, a, false, b, (short)0,
                                                      acc[j], false, false);
    }
    __syncthreads();
  }
#pragma unroll
  for (int j = 0; j < 2; ++j) {
    int n = colbase + (pair + 2 * j) * 16 + nlane;
#pragma unroll
    for (int i = 0; i < 8; ++i) {
      int row = rowbase + mtile * 16 + i + 8 * h16;
      out[(size_t)row * 4096 + n] = acc[j][i];
    }
  }
}

// W[K][N] f32 -> Wt[N][KP] f16 (K-transposed, zero-padded).
__global__ void convert_wt_kernel(const float* __restrict__ W, _Float16* __restrict__ Wt,
                                  int K, int N, int KP)
{
  int idx = blockIdx.x * 256 + threadIdx.x;
  if (idx >= N * KP) return;
  int n = idx / KP, k = idx % KP;
  Wt[idx] = (k < K) ? (_Float16)W[k * N + n] : (_Float16)0.f;
}

__global__ void copy_f32v4_kernel(const float4* __restrict__ s, float4* __restrict__ d, int n4)
{
  int i = blockIdx.x * 256 + threadIdx.x;
  if (i < n4) d[i] = s[i];
}

// ---------------------------------------------------------------------------

extern "C" void kernel_launch(void* const* d_in, const int* in_sizes, int n_in,
                              void* d_out, int out_size, void* d_ws, size_t ws_size,
                              hipStream_t stream)
{
  if (n_in < 42) return;
  constexpr int NN = 4096, NE = 131072, H = 256, OUTF = 128;

  const float* node_feat = (const float*)d_in[0];
  const float* edge_feat = (const float*)d_in[1];
  const int*   src_idx   = (const int*)d_in[2];
  const int*   dst_idx   = (const int*)d_in[3];
  const float* W_node1 = (const float*)d_in[4];  const float* B_node1 = (const float*)d_in[5];
  const float* W_node2 = (const float*)d_in[6];  const float* B_node2 = (const float*)d_in[7];
  const float* W_edge1 = (const float*)d_in[8];  const float* B_edge1 = (const float*)d_in[9];
  const float* W_edge2 = (const float*)d_in[10]; const float* B_edge2 = (const float*)d_in[11];
  const float* W_out1  = (const float*)d_in[36]; const float* B_out1  = (const float*)d_in[37];
  const float* W_out2  = (const float*)d_in[38]; const float* B_out2  = (const float*)d_in[39];
  const float* W_enc   = (const float*)d_in[40]; const float* B_enc   = (const float*)d_in[41];

  // ---- workspace layout (bump allocator, 256B aligned) ----
  char* ws = (char*)d_ws;
  size_t off = 0;
  auto alloc = [&](size_t bytes) -> char* {
    char* p = ws + off; off += (bytes + 255) & ~(size_t)255; return p;
  };
  float*    h_f32    = (float*)alloc((size_t)NN * H * 4);
  float*    agg      = (float*)alloc((size_t)NN * H * 4);
  _Float16* h_f16    = (_Float16*)alloc((size_t)NN * H * 2);
  _Float16* hf_f16   = (_Float16*)alloc((size_t)NN * OUTF * 2);
  _Float16* henc_f16 = (_Float16*)alloc((size_t)NN * OUTF * 2);
  _Float16* Wt_node1 = (_Float16*)alloc((size_t)H * 32 * 2);
  _Float16* Wt_node2 = (_Float16*)alloc((size_t)H * H * 2);
  _Float16* Wt_edge1 = (_Float16*)alloc((size_t)H * 32 * 2);
  _Float16* Wt_edge2 = (_Float16*)alloc((size_t)H * H * 2);
  _Float16* Wt_m1l1[3]; _Float16* Wt_m1l2[3]; _Float16* Wt_m2l1[3]; _Float16* Wt_m2l2[3];
  for (int L = 0; L < 3; ++L) {
    Wt_m1l1[L] = (_Float16*)alloc((size_t)H * 512 * 2);
    Wt_m1l2[L] = (_Float16*)alloc((size_t)H * H * 2);
    Wt_m2l1[L] = (_Float16*)alloc((size_t)H * H * 2);
    Wt_m2l2[L] = (_Float16*)alloc((size_t)H * H * 2);
  }
  _Float16* Wt_out1 = (_Float16*)alloc((size_t)OUTF * H * 2);
  _Float16* Wt_out2 = (_Float16*)alloc((size_t)OUTF * OUTF * 2);
  _Float16* Wt_enc  = (_Float16*)alloc((size_t)OUTF * OUTF * 2);
  _Float16* e_f16   = (_Float16*)alloc((size_t)NE * H * 2);
  if (off > ws_size) return;   // not enough scratch; bail deterministically

  auto conv = [&](const float* W, _Float16* Wt, int K, int N, int KP) {
    int total = N * KP;
    convert_wt_kernel<<<(total + 255) / 256, 256, 0, stream>>>(W, Wt, K, N, KP);
  };

  // ---- one-time (per launch) weight conversion to transposed f16 ----
  conv(W_node1, Wt_node1, 16, H, 32);
  conv(W_node2, Wt_node2, H, H, H);
  conv(W_edge1, Wt_edge1, 8, H, 32);
  conv(W_edge2, Wt_edge2, H, H, H);
  for (int L = 0; L < 3; ++L) {
    const float* w0 = (const float*)d_in[12 + 8 * L + 0];
    const float* w1 = (const float*)d_in[12 + 8 * L + 2];
    const float* w2 = (const float*)d_in[12 + 8 * L + 4];
    const float* w3 = (const float*)d_in[12 + 8 * L + 6];
    conv(w0, Wt_m1l1[L], 512, H, 512);
    conv(w1, Wt_m1l2[L], H, H, H);
    conv(w2, Wt_m2l1[L], H, H, H);
    conv(w3, Wt_m2l2[L], H, H, H);
  }
  conv(W_out1, Wt_out1, H, OUTF, H);
  conv(W_out2, Wt_out2, OUTF, OUTF, OUTF);
  conv(W_enc,  Wt_enc,  OUTF, OUTF, OUTF);

  const int node_blocks = NN / 64;   // 64
  const int edge_blocks = NE / 64;   // 2048

  // ---- encoders ----
  mlp2_kernel<MODE_NODE_ENC><<<node_blocks, 256, 0, stream>>>(
      node_feat, nullptr, nullptr, nullptr, nullptr, nullptr,
      Wt_node1, B_node1, Wt_node2, B_node2, h_f32, h_f16, nullptr, NN);
  mlp2_kernel<MODE_EDGE_ENC><<<edge_blocks, 256, 0, stream>>>(
      edge_feat, nullptr, nullptr, nullptr, nullptr, nullptr,
      Wt_edge1, B_edge1, Wt_edge2, B_edge2, nullptr, e_f16, nullptr, NE);

  // ---- 3 message-passing layers ----
  for (int L = 0; L < 3; ++L) {
    const float* b_m1l1 = (const float*)d_in[12 + 8 * L + 1];
    const float* b_m1l2 = (const float*)d_in[12 + 8 * L + 3];
    const float* b_m2l1 = (const float*)d_in[12 + 8 * L + 5];
    const float* b_m2l2 = (const float*)d_in[12 + 8 * L + 7];

    // agg = h  (segment_sum lands on top of this via atomics)
    copy_f32v4_kernel<<<(NN * H / 4 + 255) / 256, 256, 0, stream>>>(
        (const float4*)h_f32, (float4*)agg, NN * H / 4);

    // m = MLP(cat(h[src], e)); agg[dst] += m   (fused, atomic scatter)
    mlp2_kernel<MODE_MSG><<<edge_blocks, 256, 0, stream>>>(
        nullptr, nullptr, h_f16, e_f16, src_idx, dst_idx,
        Wt_m1l1[L], b_m1l1, Wt_m1l2[L], b_m1l2, nullptr, nullptr, agg, NE);

    // h = MLP(agg) + h (residual), refresh f32 + f16 copies
    mlp2_kernel<MODE_NODE_UPD><<<node_blocks, 256, 0, stream>>>(
        agg, nullptr, nullptr, nullptr, nullptr, nullptr,
        Wt_m2l1[L], b_m2l1, Wt_m2l2[L], b_m2l2, h_f32, h_f16, nullptr, NN);
  }

  // ---- output MLP: hf = MLP_out(h)  [4096,128] ----
  mlp2_kernel<MODE_OUT><<<node_blocks, 256, 0, stream>>>(
      nullptr, h_f16, nullptr, nullptr, nullptr, nullptr,
      Wt_out1, B_out1, Wt_out2, B_out2, nullptr, hf_f16, nullptr, NN);

  // ---- encoder linear: henc = hf @ W_enc + b_enc ----
  linear_kernel<128, 128><<<node_blocks, 256, 0, stream>>>(
      hf_f16, Wt_enc, B_enc, henc_f16, NN);

  // ---- out = henc @ hf^T  [4096,4096,1] ----
  dim3 grid(NN / 64, NN / 64);
  hht_kernel<<<grid, 256, 0, stream>>>(henc_f16, hf_f16, (float*)d_out);
}
